// WqAqBFP16FP16Linear_32968168964171
// MI455X (gfx1250) — compile-verified, matrix operand order checked by reading, not run on hardware
//
#include <hip/hip_runtime.h>
#include <hip/hip_fp16.h>
#include <stdint.h>

typedef _Float16 v8h  __attribute__((ext_vector_type(8)));
typedef _Float16 v16h __attribute__((ext_vector_type(16)));
typedef float    v8f  __attribute__((ext_vector_type(8)));

#define M_DIM 8192
#define N_DIM 16384
#define K_DIM 4096
#define BM 128
#define BN 256
#define BK 64
#define PBK 72   // padded LDS row stride in halves (144 B): banks 36*row mod 64 all distinct

// fallback tile config (round-1 kernel)
#define FBM 128
#define FBN 128

// ---------------------------------------------------------------------------
// Pre-conversion kernels (memory-bound, ~16 us total at 23.3 TB/s)
// ---------------------------------------------------------------------------
__global__ __launch_bounds__(256) void convert_x_f32_to_f16(
    const float* __restrict__ in, _Float16* __restrict__ outp)
{
    const size_t i = ((size_t)blockIdx.x * 256 + threadIdx.x) * 8;
    float4 a = *reinterpret_cast<const float4*>(in + i);
    float4 b = *reinterpret_cast<const float4*>(in + i + 4);
    v8h h;
    h[0] = (_Float16)a.x; h[1] = (_Float16)a.y; h[2] = (_Float16)a.z; h[3] = (_Float16)a.w;
    h[4] = (_Float16)b.x; h[5] = (_Float16)b.y; h[6] = (_Float16)b.z; h[7] = (_Float16)b.w;
    *reinterpret_cast<v8h*>(outp + i) = h;
}

__global__ __launch_bounds__(256) void convert_w_i8_to_f16(
    const int8_t* __restrict__ in, _Float16* __restrict__ outp)
{
    const size_t i = ((size_t)blockIdx.x * 256 + threadIdx.x) * 16;
    int4 raw = *reinterpret_cast<const int4*>(in + i);
    const int8_t* wb = reinterpret_cast<const int8_t*>(&raw);
    v8h h0, h1;
#pragma unroll
    for (int j = 0; j < 8; ++j) {
        h0[j] = (_Float16)(short)wb[j];
        h1[j] = (_Float16)(short)wb[8 + j];
    }
    v8h* dst = reinterpret_cast<v8h*>(outp + i);
    dst[0] = h0;
    dst[1] = h1;
}

// ---------------------------------------------------------------------------
// Main GEMM: 128x256 block tile, 4x4 WMMA tiles per wave, fp16 operands,
// double-buffered LDS filled by async global->LDS DMA.
// ---------------------------------------------------------------------------
__global__ __launch_bounds__(256) void gemm_wmma_f16_async(
    const _Float16* __restrict__ xh,    // [M, K] fp16
    const _Float16* __restrict__ wh,    // [N, K] fp16
    const _Float16* __restrict__ bias,  // [N] fp16
    const float* __restrict__ alpha_p,  // scalar
    float* __restrict__ out)            // [M, N] fp32
{
    __shared__ __align__(16) _Float16 As[2][BM * PBK];   // 2 x 18 KB
    __shared__ __align__(16) _Float16 Bs[2][BN * PBK];   // 2 x 36 KB

    const int tid  = threadIdx.x;          // 0..255
    const int lane = tid & 31;
    const int wid  = tid >> 5;             // 0..7
    const int l16  = lane & 15;
    const int sel  = lane >> 4;

    const int n0 = blockIdx.x * BN;
    const int m0 = blockIdx.y * BM;

    const int wrow = wid >> 2;             // 0..1 -> 64 M-rows each
    const int wcol = wid & 3;              // 0..3 -> 64 N-cols each

    v8f acc[4][4];
#pragma unroll
    for (int mi = 0; mi < 4; ++mi)
#pragma unroll
        for (int ni = 0; ni < 4; ++ni)
#pragma unroll
            for (int r = 0; r < 8; ++r)
                acc[mi][ni][r] = 0.0f;

    // A staging: 2 threads per row, 64 B (32 halves) each
    const int arow = tid >> 1;             // 0..127
    const int aseg = tid & 1;              // 0,1
    // B staging: 1 thread per row, 128 B each
    const int brow = tid;                  // 0..255

    const _Float16* xg = xh + (size_t)(m0 + arow) * K_DIM + aseg * 32;
    const _Float16* wg = wh + (size_t)(n0 + brow) * K_DIM;

    // LDS byte offsets (flat->local addrspacecast truncates to low 32 bits)
    uint32_t lAs[2], lBs[2];
    lAs[0] = (uint32_t)(uintptr_t)&As[0][arow * PBK + aseg * 32];
    lAs[1] = (uint32_t)(uintptr_t)&As[1][arow * PBK + aseg * 32];
    lBs[0] = (uint32_t)(uintptr_t)&Bs[0][brow * PBK];
    lBs[1] = (uint32_t)(uintptr_t)&Bs[1][brow * PBK];

    // async-stage one BK tile pair into LDS buffer `buf`
    auto stage = [&](int buf, int k0) {
        const _Float16* gx = xg + k0;
        const _Float16* gw = wg + k0;
        asm volatile(
            "global_load_async_to_lds_b128 %0, %1, off\n\t"
            "global_load_async_to_lds_b128 %0, %1, off offset:16\n\t"
            "global_load_async_to_lds_b128 %0, %1, off offset:32\n\t"
            "global_load_async_to_lds_b128 %0, %1, off offset:48"
            :
            : "v"(lAs[buf]), "v"(gx)
            : "memory");
        asm volatile(
            "global_load_async_to_lds_b128 %0, %1, off\n\t"
            "global_load_async_to_lds_b128 %0, %1, off offset:16\n\t"
            "global_load_async_to_lds_b128 %0, %1, off offset:32\n\t"
            "global_load_async_to_lds_b128 %0, %1, off offset:48\n\t"
            "global_load_async_to_lds_b128 %0, %1, off offset:64\n\t"
            "global_load_async_to_lds_b128 %0, %1, off offset:80\n\t"
            "global_load_async_to_lds_b128 %0, %1, off offset:96\n\t"
            "global_load_async_to_lds_b128 %0, %1, off offset:112"
            :
            : "v"(lBs[buf]), "v"(gw)
            : "memory");
    };

    auto compute = [&](int buf) {
#pragma unroll
        for (int kk = 0; kk < BK; kk += 32) {
            v16h afrag[4];
#pragma unroll
            for (int mi = 0; mi < 4; ++mi) {
                const v8h* ar = reinterpret_cast<const v8h*>(
                    &As[buf][(wrow * 64 + mi * 16 + l16) * PBK + kk]);
                v8h lo = ar[sel];        // K chunk: sel ? [8..15] : [0..7]
                v8h hi = ar[2 + sel];    // K chunk: sel ? [24..31] : [16..23]
                afrag[mi] = __builtin_shufflevector(lo, hi,
                    0,1,2,3,4,5,6,7,8,9,10,11,12,13,14,15);
            }
#pragma unroll
            for (int ni = 0; ni < 4; ++ni) {
                const v8h* br = reinterpret_cast<const v8h*>(
                    &Bs[buf][(wcol * 64 + ni * 16 + l16) * PBK + kk]);
                v8h lo = br[sel];
                v8h hi = br[2 + sel];
                v16h bfrag = __builtin_shufflevector(lo, hi,
                    0,1,2,3,4,5,6,7,8,9,10,11,12,13,14,15);
#pragma unroll
                for (int mi = 0; mi < 4; ++mi) {
                    acc[mi][ni] = __builtin_amdgcn_wmma_f32_16x16x32_f16(
                        false, afrag[mi], false, bfrag,
                        (short)0, acc[mi][ni], false, false);
                }
            }
        }
    };

    // prologue: fill buffer 0
    stage(0, 0);
    asm volatile("s_wait_asynccnt 0x0" ::: "memory");
    __syncthreads();

    int buf = 0;
    for (int k0 = 0; k0 < K_DIM; k0 += BK) {
        const int nbuf = buf ^ 1;
        if (k0 + BK < K_DIM) stage(nbuf, k0 + BK);   // overlap DMA with compute
        compute(buf);
        asm volatile("s_wait_asynccnt 0x0" ::: "memory");
        __syncthreads();
        buf = nbuf;
    }

    // epilogue: alpha scale + fp16 bias, per C/D VGPR layout
    const float alpha = alpha_p[0];
#pragma unroll
    for (int ni = 0; ni < 4; ++ni) {
        const int gn = n0 + wcol * 64 + ni * 16 + l16;
        const float bv = (float)bias[gn];
#pragma unroll
        for (int mi = 0; mi < 4; ++mi) {
#pragma unroll
            for (int r = 0; r < 8; ++r) {
                const int gm = m0 + wrow * 64 + mi * 16 + sel * 8 + r;
                out[(size_t)gm * N_DIM + gn] = acc[mi][ni][r] * alpha + bv;
            }
        }
    }
}

// ---------------------------------------------------------------------------
// Fallback: inline-conversion GEMM (round-1 kernel) if workspace is too small
// ---------------------------------------------------------------------------
__global__ __launch_bounds__(256) void gemm_wmma_f16_inline(
    const float* __restrict__ x,
    const int8_t* __restrict__ w,
    const _Float16* __restrict__ bias,
    const float* __restrict__ alpha_p,
    float* __restrict__ out)
{
    __shared__ __align__(16) _Float16 As[FBM * BK];
    __shared__ __align__(16) _Float16 Bs[FBN * BK];

    const int tid  = threadIdx.x;
    const int lane = tid & 31;
    const int wid  = tid >> 5;
    const int l16  = lane & 15;
    const int sel  = lane >> 4;

    const int n0 = blockIdx.x * FBN;
    const int m0 = blockIdx.y * FBM;

    const int wrow = wid >> 1;
    const int wcol = wid & 1;

    v8f acc[2][4];
#pragma unroll
    for (int mi = 0; mi < 2; ++mi)
#pragma unroll
        for (int ni = 0; ni < 4; ++ni)
#pragma unroll
            for (int r = 0; r < 8; ++r)
                acc[mi][ni][r] = 0.0f;

    const int srow = tid >> 2;
    const int scol = (tid & 3) * 16;

    for (int k0 = 0; k0 < K_DIM; k0 += BK) {
#pragma unroll
        for (int p = 0; p < 2; ++p) {
            const int row = p * 64 + srow;
            const float4* src = reinterpret_cast<const float4*>(
                x + (size_t)(m0 + row) * K_DIM + (k0 + scol));
            float fv[16];
            *reinterpret_cast<float4*>(&fv[0])  = src[0];
            *reinterpret_cast<float4*>(&fv[4])  = src[1];
            *reinterpret_cast<float4*>(&fv[8])  = src[2];
            *reinterpret_cast<float4*>(&fv[12]) = src[3];
            v8h h0, h1;
#pragma unroll
            for (int i = 0; i < 8; ++i) {
                h0[i] = (_Float16)fv[i];
                h1[i] = (_Float16)fv[8 + i];
            }
            v8h* dst = reinterpret_cast<v8h*>(&As[row * BK + scol]);
            dst[0] = h0;
            dst[1] = h1;
        }
#pragma unroll
        for (int p = 0; p < 2; ++p) {
            const int row = p * 64 + srow;
            int4 raw = *reinterpret_cast<const int4*>(
                w + (size_t)(n0 + row) * K_DIM + (k0 + scol));
            const int8_t* wb = reinterpret_cast<const int8_t*>(&raw);
            v8h h0, h1;
#pragma unroll
            for (int i = 0; i < 8; ++i) {
                h0[i] = (_Float16)(short)wb[i];
                h1[i] = (_Float16)(short)wb[8 + i];
            }
            v8h* dst = reinterpret_cast<v8h*>(&Bs[row * BK + scol]);
            dst[0] = h0;
            dst[1] = h1;
        }
        __syncthreads();

#pragma unroll
        for (int kk = 0; kk < BK; kk += 32) {
            v16h afrag[2];
#pragma unroll
            for (int mi = 0; mi < 2; ++mi) {
                const v8h* ar = reinterpret_cast<const v8h*>(
                    &As[(wrow * 32 + mi * 16 + l16) * BK + kk]);
                v8h lo = ar[sel];
                v8h hi = ar[2 + sel];
                afrag[mi] = __builtin_shufflevector(lo, hi,
                    0,1,2,3,4,5,6,7,8,9,10,11,12,13,14,15);
            }
#pragma unroll
            for (int ni = 0; ni < 4; ++ni) {
                const v8h* br = reinterpret_cast<const v8h*>(
                    &Bs[(wcol * 64 + ni * 16 + l16) * BK + kk]);
                v8h lo = br[sel];
                v8h hi = br[2 + sel];
                v16h bfrag = __builtin_shufflevector(lo, hi,
                    0,1,2,3,4,5,6,7,8,9,10,11,12,13,14,15);
#pragma unroll
                for (int mi = 0; mi < 2; ++mi) {
                    acc[mi][ni] = __builtin_amdgcn_wmma_f32_16x16x32_f16(
                        false, afrag[mi], false, bfrag,
                        (short)0, acc[mi][ni], false, false);
                }
            }
        }
        __syncthreads();
    }

    const float alpha = alpha_p[0];
#pragma unroll
    for (int ni = 0; ni < 4; ++ni) {
        const int gn = n0 + wcol * 64 + ni * 16 + l16;
        const float bv = (float)bias[gn];
#pragma unroll
        for (int mi = 0; mi < 2; ++mi) {
#pragma unroll
            for (int r = 0; r < 8; ++r) {
                const int gm = m0 + wrow * 32 + mi * 16 + sel * 8 + r;
                out[(size_t)gm * N_DIM + gn] = acc[mi][ni][r] * alpha + bv;
            }
        }
    }
}

extern "C" void kernel_launch(void* const* d_in, const int* in_sizes, int n_in,
                              void* d_out, int out_size, void* d_ws, size_t ws_size,
                              hipStream_t stream) {
    (void)in_sizes; (void)n_in; (void)out_size;
    const float*    x     = (const float*)d_in[0];
    const int8_t*   w     = (const int8_t*)d_in[1];
    const _Float16* bias  = (const _Float16*)d_in[2];
    const float*    alpha = (const float*)d_in[3];
    float*          out   = (float*)d_out;

    const size_t xh_bytes = (size_t)M_DIM * K_DIM * sizeof(_Float16);  // 64 MB
    const size_t wh_bytes = (size_t)N_DIM * K_DIM * sizeof(_Float16);  // 128 MB

    dim3 block(256);

    if (ws_size >= xh_bytes + wh_bytes) {
        _Float16* xh = (_Float16*)d_ws;
        _Float16* wh = (_Float16*)((char*)d_ws + xh_bytes);
        convert_x_f32_to_f16<<<dim3((M_DIM * (size_t)K_DIM) / (256 * 8)), block, 0, stream>>>(x, xh);
        convert_w_i8_to_f16<<<dim3((N_DIM * (size_t)K_DIM) / (256 * 16)), block, 0, stream>>>(w, wh);
        dim3 grid(N_DIM / BN, M_DIM / BM);   // (64, 64)
        gemm_wmma_f16_async<<<grid, block, 0, stream>>>(xh, wh, bias, alpha, out);
    } else {
        dim3 grid(N_DIM / FBN, M_DIM / FBM); // (128, 64)
        gemm_wmma_f16_inline<<<grid, block, 0, stream>>>(x, w, bias, alpha, out);
    }
}
